// HeadAttention_33363305955709
// MI455X (gfx1250) — matrix operand order repaired and checked by reading, not compiled, not run on hardware
//
#include <hip/hip_runtime.h>
#include <math.h>

#define EMBED 1024
#define SEQ   2048
#define NB    4

typedef __attribute__((ext_vector_type(16))) _Float16       v16h;
typedef __attribute__((ext_vector_type(8)))  float          v8f;
typedef __attribute__((ext_vector_type(8)))  unsigned short u16x8;
typedef __attribute__((ext_vector_type(8)))  _Float16       h8;
typedef __attribute__((ext_vector_type(4)))  _Float16       h4;
typedef __attribute__((ext_vector_type(4)))  float          f4;

// Matches the builtin's parameter pointee type exactly (GCC vector of 4 ints).
typedef int i4v __attribute__((vector_size(16)));

#if __has_builtin(__builtin_amdgcn_global_load_async_to_lds_b128) && \
    __has_builtin(__builtin_amdgcn_s_wait_asynccnt)
#define HAVE_ASYNC_LDS 1
#else
#define HAVE_ASYNC_LDS 0
#endif

// 16-byte global -> LDS copy. Async (ASYNCcnt-tracked, no VGPR round trip)
// when the gfx1250 builtin is available, plain copy otherwise.
static __device__ __forceinline__ void cp16_to_lds(_Float16* lds_dst, const _Float16* gsrc) {
#if HAVE_ASYNC_LDS
  __builtin_amdgcn_global_load_async_to_lds_b128(
      (__attribute__((address_space(1))) i4v*)(size_t)gsrc,
      (__attribute__((address_space(3))) i4v*)(unsigned)(size_t)lds_dst,
      0, 0);
#else
  *(h8*)lds_dst = *(const h8*)gsrc;
#endif
}

static __device__ __forceinline__ void wait_stage_and_barrier() {
#if HAVE_ASYNC_LDS
  __builtin_amdgcn_s_wait_asynccnt(0);
#endif
  __syncthreads();
}

union FragU { v16h v; u16x8 h[2]; };

// A-matrix fragment (16x32 f16, M=lane%16):
//   elems 0..7  : K = (lane>=16? 8:0) + 0..7
//   elems 8..15 : K = 16 + (lane>=16? 8:0) + 0..7
static __device__ __forceinline__ v16h lds_frag_A(const _Float16* tile, int row, int lane) {
  const int hs = (lane >> 4) & 1;
  const _Float16* p = tile + row * 32;
  FragU f;
  f.h[0] = *(const u16x8*)(p + hs * 8);
  f.h[1] = *(const u16x8*)(p + 16 + hs * 8);
  return f.v;
}

// B-matrix fragment (32x16 f16, N=lane%16): elems 0..7 = K = (lane>=16?8:0)+0..7,
// elems 8..15 = K = 16 + (lane>=16?8:0)+0..7.
// Source tile is row-major [n][k] (rows of K / W / V^T).
static __device__ __forceinline__ v16h lds_frag_B(const _Float16* tile, int row, int lane) {
  const int hs = (lane >> 4) & 1;
  const _Float16* p = tile + row * 32;
  FragU f;
  f.h[0] = *(const u16x8*)(p + hs * 8);
  f.h[1] = *(const u16x8*)(p + 16 + hs * 8);
  return f.v;
}

// ---------------------------------------------------------------------------
// Kernel 1: Y[m][n] = sum_d X[m][d] * W[n][d] + b[n]   (fp32 in, f16 out)
// Block tile 128(M) x 128(N), 8 waves -> 32x64 per wave = 2x4 WMMA tiles.
// fp32 -> f16 conversion happens during LDS staging.
// ---------------------------------------------------------------------------
__global__ __launch_bounds__(256) void proj_kernel(const float* __restrict__ X,
                                                   const float* __restrict__ W,
                                                   const float* __restrict__ bias,
                                                   _Float16* __restrict__ Y) {
  __shared__ __align__(16) _Float16 sA[128 * 32];
  __shared__ __align__(16) _Float16 sB[128 * 32];
  const int tid  = threadIdx.x;
  const int lane = tid & 31;
  const int wid  = tid >> 5;
  const int mw   = wid & 3;   // 4 M sub-tiles of 32
  const int nw   = wid >> 2;  // 2 N sub-tiles of 64
  const int m0   = blockIdx.y * 128;
  const int n0   = blockIdx.x * 128;

  v8f acc[2][4] = {};

  for (int k0 = 0; k0 < EMBED; k0 += 32) {
    __syncthreads();
    #pragma unroll
    for (int i = 0; i < 4; ++i) {
      int idx = i * 1024 + tid * 4;
      int row = idx >> 5, kk = idx & 31;
      const float* gp = X + (size_t)(m0 + row) * EMBED + k0 + kk;
      f4 xv = *(const f4*)gp;
      __builtin_prefetch((const void*)(gp + 32), 0, 0);  // speculative: safe past end
      h4 hv;
      hv[0] = (_Float16)xv[0]; hv[1] = (_Float16)xv[1];
      hv[2] = (_Float16)xv[2]; hv[3] = (_Float16)xv[3];
      *(h4*)(sA + idx) = hv;
    }
    #pragma unroll
    for (int i = 0; i < 4; ++i) {
      int idx = i * 1024 + tid * 4;
      int row = idx >> 5, kk = idx & 31;
      f4 wv = *(const f4*)(W + (size_t)(n0 + row) * EMBED + k0 + kk);
      h4 hv;
      hv[0] = (_Float16)wv[0]; hv[1] = (_Float16)wv[1];
      hv[2] = (_Float16)wv[2]; hv[3] = (_Float16)wv[3];
      *(h4*)(sB + idx) = hv;
    }
    __syncthreads();

    v16h a[2], b[4];
    #pragma unroll
    for (int tm = 0; tm < 2; ++tm) a[tm] = lds_frag_A(sA, mw * 32 + tm * 16 + (lane & 15), lane);
    #pragma unroll
    for (int tn = 0; tn < 4; ++tn) b[tn] = lds_frag_B(sB, nw * 64 + tn * 16 + (lane & 15), lane);
    #pragma unroll
    for (int tm = 0; tm < 2; ++tm)
      #pragma unroll
      for (int tn = 0; tn < 4; ++tn)
        acc[tm][tn] = __builtin_amdgcn_wmma_f32_16x16x32_f16(
            false, a[tm], false, b[tn], (short)0, acc[tm][tn], false, false);
  }

  const int hs8 = ((lane >> 4) & 1) * 8;
  #pragma unroll
  for (int tm = 0; tm < 2; ++tm) {
    #pragma unroll
    for (int tn = 0; tn < 4; ++tn) {
      const int ng = n0 + nw * 64 + tn * 16 + (lane & 15);
      const float bv = bias[ng];
      #pragma unroll
      for (int r = 0; r < 8; ++r) {
        const int mg = m0 + mw * 32 + tm * 16 + hs8 + r;
        Y[(size_t)mg * EMBED + ng] = (_Float16)(acc[tm][tn][r] + bv);
      }
    }
  }
}

// ---------------------------------------------------------------------------
// Kernel 2: S[b][q][kv] = scale * dot(Q[b][q], K[b][kv]), causal mask -> -inf
// f16 in/out. Block tile 128(q) x 128(kv), double-buffered async LDS staging.
// ---------------------------------------------------------------------------
__global__ __launch_bounds__(256) void scores_kernel(const _Float16* __restrict__ Qh,
                                                     const _Float16* __restrict__ Kh,
                                                     _Float16* __restrict__ Sh,
                                                     float scale) {
  __shared__ __align__(16) _Float16 sQ[2][128 * 32];
  __shared__ __align__(16) _Float16 sK[2][128 * 32];
  const int tid  = threadIdx.x;
  const int lane = tid & 31;
  const int wid  = tid >> 5;
  const int mw   = wid & 3;
  const int nw   = wid >> 2;
  const int b    = blockIdx.z;
  const int q0   = blockIdx.y * 128;
  const int kv0  = blockIdx.x * 128;
  _Float16* srow = Sh + (size_t)b * SEQ * SEQ;

  if (kv0 > q0 + 127) {  // fully masked tile (uniform branch)
    h8 ni;
    #pragma unroll
    for (int j = 0; j < 8; ++j) ni[j] = (_Float16)(-INFINITY);
    #pragma unroll
    for (int i = 0; i < 8; ++i) {
      int idx = i * 2048 + tid * 8;
      int rq = idx >> 7, ck = idx & 127;
      *(h8*)(srow + (size_t)(q0 + rq) * SEQ + kv0 + ck) = ni;
    }
    return;
  }

  const _Float16* Qb = Qh + (size_t)b * SEQ * EMBED;
  const _Float16* Kb = Kh + (size_t)b * SEQ * EMBED;

  auto stage = [&](int buf, int k0) {
    #pragma unroll
    for (int i = 0; i < 2; ++i) {
      int idx = i * 2048 + tid * 8;
      int row = idx >> 5, kk = idx & 31;
      cp16_to_lds(&sQ[buf][idx], Qb + (size_t)(q0 + row) * EMBED + k0 + kk);
      cp16_to_lds(&sK[buf][idx], Kb + (size_t)(kv0 + row) * EMBED + k0 + kk);
    }
  };

  v8f acc[2][4] = {};
  stage(0, 0);
  wait_stage_and_barrier();

  for (int k0 = 0; k0 < EMBED; k0 += 32) {
    const int cur = (k0 >> 5) & 1;
    if (k0 + 32 < EMBED) stage(cur ^ 1, k0 + 32);  // fill next buffer while computing

    v16h a[2], bb[4];
    #pragma unroll
    for (int tm = 0; tm < 2; ++tm) a[tm] = lds_frag_A(sQ[cur], mw * 32 + tm * 16 + (lane & 15), lane);
    #pragma unroll
    for (int tn = 0; tn < 4; ++tn) bb[tn] = lds_frag_B(sK[cur], nw * 64 + tn * 16 + (lane & 15), lane);
    #pragma unroll
    for (int tm = 0; tm < 2; ++tm)
      #pragma unroll
      for (int tn = 0; tn < 4; ++tn)
        acc[tm][tn] = __builtin_amdgcn_wmma_f32_16x16x32_f16(
            false, a[tm], false, bb[tn], (short)0, acc[tm][tn], false, false);

    wait_stage_and_barrier();
  }

  const bool full = (kv0 + 127 <= q0);  // no element masked
  const int hs8 = ((lane >> 4) & 1) * 8;
  #pragma unroll
  for (int tm = 0; tm < 2; ++tm) {
    #pragma unroll
    for (int tn = 0; tn < 4; ++tn) {
      const int kg = kv0 + nw * 64 + tn * 16 + (lane & 15);
      #pragma unroll
      for (int r = 0; r < 8; ++r) {
        const int qg = q0 + mw * 32 + tm * 16 + hs8 + r;
        float v = acc[tm][tn][r] * scale;
        if (!full && kg > qg) v = -INFINITY;
        srow[(size_t)qg * SEQ + kg] = (_Float16)v;
      }
    }
  }
}

// ---------------------------------------------------------------------------
// Kernel 3: row softmax, in place (f16 -> f16). One 256-thread block per row.
// ---------------------------------------------------------------------------
__global__ __launch_bounds__(256) void softmax_kernel(_Float16* __restrict__ Sh) {
  __shared__ float red[8];
  const int tid  = threadIdx.x;
  const int lane = tid & 31;
  const int wid  = tid >> 5;
  _Float16* p = Sh + (size_t)blockIdx.x * SEQ;

  float v[8];
  #pragma unroll
  for (int j = 0; j < 8; ++j) v[j] = (float)p[tid + j * 256];

  float m = -INFINITY;
  #pragma unroll
  for (int j = 0; j < 8; ++j) m = fmaxf(m, v[j]);
  #pragma unroll
  for (int off = 16; off > 0; off >>= 1) m = fmaxf(m, __shfl_xor(m, off));
  if (lane == 0) red[wid] = m;
  __syncthreads();
  m = red[0];
  #pragma unroll
  for (int i = 1; i < 8; ++i) m = fmaxf(m, red[i]);

  float e[8], s = 0.0f;
  #pragma unroll
  for (int j = 0; j < 8; ++j) { e[j] = __expf(v[j] - m); s += e[j]; }
  #pragma unroll
  for (int off = 16; off > 0; off >>= 1) s += __shfl_xor(s, off);
  __syncthreads();
  if (lane == 0) red[wid] = s;
  __syncthreads();
  s = 0.0f;
  #pragma unroll
  for (int i = 0; i < 8; ++i) s += red[i];

  const float rinv = 1.0f / s;
  #pragma unroll
  for (int j = 0; j < 8; ++j) p[tid + j * 256] = (_Float16)(e[j] * rinv);
}

// ---------------------------------------------------------------------------
// Kernel 4: O[b][q][e] = sum_kv P[b][q][kv] * V[b][kv][e]   (f16 in, fp32 out)
// P staged async; V tile transposed into LDS so B-fragments are contiguous-K.
// Causal: P is zero for kv > q, so K-loop stops at q0+128.
// ---------------------------------------------------------------------------
__global__ __launch_bounds__(256) void pv_kernel(const _Float16* __restrict__ Ph,
                                                 const _Float16* __restrict__ Vh,
                                                 float* __restrict__ O) {
  __shared__ __align__(16) _Float16 sP[2][128 * 32];
  __shared__ __align__(16) _Float16 sVT[2][128 * 32];
  const int tid  = threadIdx.x;
  const int lane = tid & 31;
  const int wid  = tid >> 5;
  const int mw   = wid & 3;
  const int nw   = wid >> 2;
  const int b    = blockIdx.z;
  const int q0   = blockIdx.y * 128;
  const int n0   = blockIdx.x * 128;
  const _Float16* Pb = Ph + (size_t)b * SEQ * SEQ;
  const _Float16* Vb = Vh + (size_t)b * SEQ * EMBED;

  auto stage = [&](int buf, int k0) {
    #pragma unroll
    for (int i = 0; i < 2; ++i) {          // P tile 128x32, async copy
      int idx = i * 2048 + tid * 8;
      int row = idx >> 5, kk = idx & 31;
      cp16_to_lds(&sP[buf][idx], Pb + (size_t)(q0 + row) * SEQ + k0 + kk);
    }
    #pragma unroll
    for (int i = 0; i < 2; ++i) {          // V tile 32(k) x 128(n), transposed
      int linear = i * 2048 + tid * 8;
      int kr = linear >> 7, nc = linear & 127;
      h8 vv = *(const h8*)(Vb + (size_t)(k0 + kr) * EMBED + n0 + nc);
      #pragma unroll
      for (int j = 0; j < 8; ++j) sVT[buf][(nc + j) * 32 + kr] = vv[j];
    }
  };

  v8f acc[2][4] = {};
  const int kend = (q0 + 128 < SEQ) ? (q0 + 128) : SEQ;

  stage(0, 0);
  wait_stage_and_barrier();

  for (int k0 = 0; k0 < kend; k0 += 32) {
    const int cur = (k0 >> 5) & 1;
    if (k0 + 32 < kend) stage(cur ^ 1, k0 + 32);

    v16h a[2], bb[4];
    #pragma unroll
    for (int tm = 0; tm < 2; ++tm) a[tm] = lds_frag_A(sP[cur], mw * 32 + tm * 16 + (lane & 15), lane);
    #pragma unroll
    for (int tn = 0; tn < 4; ++tn) bb[tn] = lds_frag_B(sVT[cur], nw * 64 + tn * 16 + (lane & 15), lane);
    #pragma unroll
    for (int tm = 0; tm < 2; ++tm)
      #pragma unroll
      for (int tn = 0; tn < 4; ++tn)
        acc[tm][tn] = __builtin_amdgcn_wmma_f32_16x16x32_f16(
            false, a[tm], false, bb[tn], (short)0, acc[tm][tn], false, false);

    wait_stage_and_barrier();
  }

  const int hs8 = ((lane >> 4) & 1) * 8;
  #pragma unroll
  for (int tm = 0; tm < 2; ++tm) {
    #pragma unroll
    for (int tn = 0; tn < 4; ++tn) {
      const int ng = n0 + nw * 64 + tn * 16 + (lane & 15);
      #pragma unroll
      for (int r = 0; r < 8; ++r) {
        const int qg = q0 + mw * 32 + tm * 16 + hs8 + r;
        O[((size_t)b * SEQ + qg) * EMBED + ng] = acc[tm][tn][r];
      }
    }
  }
}

// ---------------------------------------------------------------------------
extern "C" void kernel_launch(void* const* d_in, const int* in_sizes, int n_in,
                              void* d_out, int out_size, void* d_ws, size_t ws_size,
                              hipStream_t stream) {
  const float* xq = (const float*)d_in[0];
  const float* xk = (const float*)d_in[1];
  const float* xv = (const float*)d_in[2];
  const float* Wq = (const float*)d_in[3];
  const float* bq = (const float*)d_in[4];
  // d_in[5] is the triu causal mask; causality is applied analytically.

  const size_t qkv_elems = (size_t)NB * SEQ * EMBED;   // 8M f16 each
  _Float16* qh = (_Float16*)d_ws;
  _Float16* kh = qh + qkv_elems;
  _Float16* vh = kh + qkv_elems;
  _Float16* sh = vh + qkv_elems;                       // B*S*S f16 scores/probs
  float* out = (float*)d_out;

  dim3 blk(256);
  const float scale = 1.0f / 32.0f;  // LATEN_DIM ** -0.5

  dim3 gproj(EMBED / 128, (NB * SEQ) / 128, 1);
  proj_kernel<<<gproj, blk, 0, stream>>>(xq, Wq, bq, qh);
  proj_kernel<<<gproj, blk, 0, stream>>>(xk, Wq, bq, kh);
  proj_kernel<<<gproj, blk, 0, stream>>>(xv, Wq, bq, vh);

  scores_kernel<<<dim3(SEQ / 128, SEQ / 128, NB), blk, 0, stream>>>(qh, kh, sh, scale);

  softmax_kernel<<<dim3(NB * SEQ), blk, 0, stream>>>(sh);

  pv_kernel<<<dim3(EMBED / 128, SEQ / 128, NB), blk, 0, stream>>>(sh, vh, out);
}
